// CMAModel_72722386256237
// MI455X (gfx1250) — matrix operand order repaired and checked by reading, not compiled, not run on hardware
//
#include <hip/hip_runtime.h>

// ---------------------------------------------------------------------------
// CMA (cross-memory attention) for MI455X / gfx1250, wave32.
// f16 WMMA (16x16x32, f32 accumulate) everywhere; f32 softmax/normalize.
// K tiles staged to LDS with GLOBAL_LOAD_ASYNC_TO_LDS_B128 (ASYNCcnt path).
// ---------------------------------------------------------------------------

typedef float v2f  __attribute__((ext_vector_type(2)));
typedef float v8f  __attribute__((ext_vector_type(8)));
typedef _Float16 h4  __attribute__((ext_vector_type(4)));
typedef _Float16 h8  __attribute__((ext_vector_type(8)));
typedef _Float16 v16h __attribute__((ext_vector_type(16)));

// D(16x16 f32) = A(16x32 f16) * B(32x16 f16) + C   (codegen-confirmed builtin)
#define WMMA_F16X32(a, b, c) \
  __builtin_amdgcn_wmma_f32_16x16x32_f16(false, (a), false, (b), (short)0, (c), false, false)

// Problem constants (match reference)
constexpr int Bc = 2;
constexpr int Tc = 2048;
constexpr int Cc = 768;
constexpr int Hc = 6;
constexpr int Dh = 128;            // head dim
constexpr int Mc = 3072;
constexpr int Rc = 320;
constexpr int Sc = Tc + Mc + Rc;   // 5440

__device__ __forceinline__ v16h mk16(h8 lo, h8 hi) {
  return __builtin_shufflevector(lo, hi, 0, 1, 2, 3, 4, 5, 6, 7, 8, 9, 10, 11,
                                 12, 13, 14, 15);
}
// A-fragment (16x32 f16): elements e<8 -> K = 8*hf+e ; e>=8 -> K = 16+8*hf+(e-8)
__device__ __forceinline__ v16h ldA(const _Float16* rowp, int hf) {
  h8 lo = *(const h8*)(rowp + 8 * hf);
  h8 hi = *(const h8*)(rowp + 16 + 8 * hf);
  return mk16(lo, hi);
}
// B-fragment (32x16 f16): elements e<8 -> K = 8*hf+e ; e>=8 -> K = 16+8*hf+(e-8)
__device__ __forceinline__ v16h ldB(const _Float16* rowp, int hf) {
  h8 lo = *(const h8*)(rowp + 8 * hf);
  h8 hi = *(const h8*)(rowp + 16 + 8 * hf);
  return mk16(lo, hi);
}

// ---------------------------------------------------------------------------
// Segmented row lookup over [B x (len0+len1+len2)]: (x | fwd_mem | rev_mem).
// ---------------------------------------------------------------------------
__device__ __forceinline__ const float* seg_row(
    const float* __restrict__ x0, const float* __restrict__ x1,
    const float* __restrict__ x2, int len0, int len1, int len2, int r, int K) {
  const int Sreg = len0 + len1 + len2;
  const int b = r / Sreg;
  int s = r - b * Sreg;
  if (s < len0) return x0 + ((size_t)(b * len0 + s)) * K;
  s -= len0;
  if (s < len1) return x1 + ((size_t)(b * len1 + s)) * K;
  s -= len1;
  return x2 + ((size_t)(b * len2 + s)) * K;
}

// ---------------------------------------------------------------------------
// Y = X @ W^T (+bias). 64x64 tile / 128-thread block, K-chunks of 32 staged
// into LDS as f16; f32 accumulate; output written f32 or f16 (for K/V).
// ---------------------------------------------------------------------------
__global__ __launch_bounds__(128) void gemm_xwT_64x64(
    const float* __restrict__ x0, const float* __restrict__ x1,
    const float* __restrict__ x2, int len0, int len1, int len2,
    const float* __restrict__ W, const float* __restrict__ bias,
    void* __restrict__ Yv, int out_half, int K, int N) {
  __shared__ _Float16 At[64][40];   // pitch 40 halves = 80B: aligned, low-conflict
  __shared__ _Float16 Bt[64][40];

  const int tid  = threadIdx.x;
  const int wave = tid >> 5;
  const int lane = tid & 31;
  const int hf   = lane >> 4;
  const int lr   = lane & 15;
  const int r0   = blockIdx.x * 64;
  const int n0   = blockIdx.y * 64;

  const v8f vzero = {0.f, 0.f, 0.f, 0.f, 0.f, 0.f, 0.f, 0.f};
  v8f acc[4];
#pragma unroll
  for (int j = 0; j < 4; ++j) acc[j] = vzero;

  for (int k0 = 0; k0 < K; k0 += 32) {
    __syncthreads();
    // Stage A (X rows) and B (W rows) 64x32 tiles as f16.
#pragma unroll
    for (int it = 0; it < 4; ++it) {
      const int f   = it * 128 + tid;   // float4 slot 0..511
      const int row = f >> 3;           // 0..63
      const int c4  = (f & 7) * 4;      // 0..28
      const float* xr = seg_row(x0, x1, x2, len0, len1, len2, r0 + row, K);
      float4 a4 = *(const float4*)(xr + k0 + c4);
      float4 b4 = *(const float4*)(W + ((size_t)(n0 + row)) * K + k0 + c4);
      h4 ah = {(_Float16)a4.x, (_Float16)a4.y, (_Float16)a4.z, (_Float16)a4.w};
      h4 bh = {(_Float16)b4.x, (_Float16)b4.y, (_Float16)b4.z, (_Float16)b4.w};
      *(h4*)(&At[row][c4]) = ah;
      *(h4*)(&Bt[row][c4]) = bh;
    }
    __syncthreads();

    v16h a = ldA(&At[16 * wave + lr][0], hf);
#pragma unroll
    for (int j = 0; j < 4; ++j) {
      v16h bb = ldB(&Bt[16 * j + lr][0], hf);
      acc[j] = WMMA_F16X32(a, bb, acc[j]);
    }
  }

  // C/D layout: vgpr r -> rows r (lanes0-15) / r+8 (lanes16-31); lane = N.
#pragma unroll
  for (int j = 0; j < 4; ++j) {
#pragma unroll
    for (int r = 0; r < 8; ++r) {
      const int row = r0 + 16 * wave + r + 8 * hf;
      const int col = n0 + 16 * j + lr;
      float vv = acc[j][r];
      if (bias) vv += bias[col];
      if (out_half)
        ((_Float16*)Yv)[(size_t)row * N + col] = (_Float16)vv;
      else
        ((float*)Yv)[(size_t)row * N + col] = vv;
    }
  }
}

// ---------------------------------------------------------------------------
// cw[o] = ctrl . Wc[o,:]
// ---------------------------------------------------------------------------
__global__ void ctrl_bias_kernel(const float* __restrict__ ctrl,
                                 const float* __restrict__ Wc,
                                 float* __restrict__ cw) {
  const int o = blockIdx.x * blockDim.x + threadIdx.x;
  if (o < Cc) {
    float s = 0.f;
#pragma unroll
    for (int j = 0; j < 5; ++j) s = fmaf(ctrl[j], Wc[o * 5 + j], s);
    cw[o] = s;
  }
}

// ---------------------------------------------------------------------------
// gate[row][h] = sigmoid(q[row] . Wg[h] + bg[h]); one wave per row.
// ---------------------------------------------------------------------------
__global__ __launch_bounds__(32) void gate_kernel(
    const float* __restrict__ q, const float* __restrict__ Wg,
    const float* __restrict__ bg, float* __restrict__ gate) {
  const int row  = blockIdx.x;
  const int lane = threadIdx.x;
  const float* qr = q + (size_t)row * Cc;
  for (int h = 0; h < Hc; ++h) {
    float s = 0.f;
    for (int c = lane; c < Cc; c += 32) s = fmaf(qr[c], Wg[h * Cc + c], s);
    s += __shfl_xor(s, 16);
    s += __shfl_xor(s, 8);
    s += __shfl_xor(s, 4);
    s += __shfl_xor(s, 2);
    s += __shfl_xor(s, 1);
    if (lane == 0)
      gate[(size_t)row * Hc + h] = 1.f / (1.f + __expf(-(s + bg[h])));
  }
}

// ---------------------------------------------------------------------------
// Gated flash attention. Grid (T/64, H, B); 4 waves; wave w owns query rows
// t0..t0+15. 32-key tiles: K staged to LDS via async-to-LDS (f16 copy),
// V transposed into LDS through VGPRs. Online softmax in f32; memory-path
// exp weights scaled by per-row gate before PV; denominator ungated.
// ---------------------------------------------------------------------------
__global__ __launch_bounds__(128) void flash_cma(
    const float* __restrict__ q, const _Float16* __restrict__ k,
    const _Float16* __restrict__ v, const float* __restrict__ gate,
    float* __restrict__ attn) {
  __shared__ _Float16 kt[32][136];    // [s][d]   pitch 272B
  __shared__ _Float16 vtT[128][40];   // [d][s]   pitch 80B (transposed V)
  __shared__ _Float16 pt[4][16][40];  // per-wave P tile [m][s]

  const int tid  = threadIdx.x;
  const int wave = tid >> 5;
  const int lane = tid & 31;
  const int hf   = lane >> 4;
  const int lr   = lane & 15;
  const int tq = blockIdx.x, h = blockIdx.y, b = blockIdx.z;
  const int t0 = tq * 64 + wave * 16;
  const float scale = 0.08838834764831845f;   // 1/sqrt(128)

  // Q fragments: 4 chunks of K=32, pre-scaled, converted to f16.
  v16h qa[4];
  {
    const float* qrow = q + ((size_t)(b * Tc + t0 + lr)) * Cc + h * Dh;
#pragma unroll
    for (int c = 0; c < 4; ++c) {
      const float* p0 = qrow + 32 * c + 8 * hf;
      const float* p1 = qrow + 32 * c + 16 + 8 * hf;
      v16h a;
#pragma unroll
      for (int e = 0; e < 8; ++e) {
        a[e]     = (_Float16)(p0[e] * scale);
        a[8 + e] = (_Float16)(p1[e] * scale);
      }
      qa[c] = a;
    }
  }

  const v8f vzero = {0.f, 0.f, 0.f, 0.f, 0.f, 0.f, 0.f, 0.f};
  float mrow[8], lsum[8], gr[8];
  v8f acc[8];   // 16x128 accumulator (8 d-subtiles)
#pragma unroll
  for (int r = 0; r < 8; ++r) {
    mrow[r] = -1e30f;
    lsum[r] = 0.f;
    gr[r]   = gate[((size_t)(b * Tc + t0 + r + 8 * hf)) * Hc + h];
    acc[r]  = vzero;
  }

  const int nLocal = tq * 2 + 2;               // 32-wide causal tiles touched
  const int nIter  = nLocal + (Mc + Rc) / 32;  // + 106 memory tiles

  for (int it = 0; it < nIter; ++it) {
    const bool is_mem = (it >= nLocal);
    const int  s_base = is_mem ? (Tc + (it - nLocal) * 32) : it * 32;

    __syncthreads();
    // K tile (32x128 f16 = 8KB): async copy global -> LDS, 16B per lane-op.
#pragma unroll
    for (int i = 0; i < 4; ++i) {
      const int f   = i * 128 + tid;    // 16B chunk 0..511
      const int row = f >> 4;           // 0..31
      const int c8  = (f & 15) * 8;     // half offset 0..120
      const _Float16* gp =
          k + ((size_t)(b * Sc + s_base + row)) * Cc + h * Dh + c8;
      const unsigned lds_off = (unsigned)(size_t)(&kt[row][c8]);
      asm volatile("global_load_async_to_lds_b128 %0, %1, off"
                   :: "v"(lds_off), "v"(gp)
                   : "memory");
    }
    // V tile: load f16, transpose-store into vtT[d][s].
#pragma unroll
    for (int i = 0; i < 4; ++i) {
      const int f   = i * 128 + tid;
      const int row = f >> 4;
      const int c8  = (f & 15) * 8;
      h8 vv = *(const h8*)(v + ((size_t)(b * Sc + s_base + row)) * Cc +
                           h * Dh + c8);
#pragma unroll
      for (int u = 0; u < 8; ++u) vtT[c8 + u][row] = vv[u];
    }
    asm volatile("s_wait_asynccnt 0" ::: "memory");
    __syncthreads();

    // Scores 16x32 = Q(16x128) @ K_tile^T : 4 K-chunks x 2 N-halves.
    v8f sc0 = vzero, sc1 = vzero;
#pragma unroll
    for (int c = 0; c < 4; ++c) {
      v16h b0 = ldB(&kt[lr][32 * c], hf);        // cols s_base+lr
      v16h b1 = ldB(&kt[16 + lr][32 * c], hf);   // cols s_base+16+lr
      sc0 = WMMA_F16X32(qa[c], b0, sc0);
      sc1 = WMMA_F16X32(qa[c], b1, sc1);
    }

    // Online softmax per row (row m = r + 8*hf; lane covers cols lr, 16+lr).
#pragma unroll
    for (int r = 0; r < 8; ++r) {
      float s0 = sc0[r], s1 = sc1[r];
      const int trow = t0 + r + 8 * hf;
      const bool m0 = (!is_mem) && (s_base + lr > trow);
      const bool m1 = (!is_mem) && (s_base + 16 + lr > trow);
      if (m0) s0 = -1e30f;
      if (m1) s1 = -1e30f;
      float tmax = fmaxf(s0, s1);
      tmax = fmaxf(tmax, __shfl_xor(tmax, 8));
      tmax = fmaxf(tmax, __shfl_xor(tmax, 4));
      tmax = fmaxf(tmax, __shfl_xor(tmax, 2));
      tmax = fmaxf(tmax, __shfl_xor(tmax, 1));
      const float mnew = fmaxf(mrow[r], tmax);
      const float resc = __expf(mrow[r] - mnew);
      mrow[r] = mnew;
      const float e0 = m0 ? 0.f : __expf(s0 - mnew);
      const float e1 = m1 ? 0.f : __expf(s1 - mnew);
      float es = e0 + e1;
      es += __shfl_xor(es, 8);
      es += __shfl_xor(es, 4);
      es += __shfl_xor(es, 2);
      es += __shfl_xor(es, 1);
      lsum[r] = lsum[r] * resc + es;   // ungated denominator
#pragma unroll
      for (int j = 0; j < 8; ++j) acc[j][r] *= resc;
      const float g = is_mem ? gr[r] : 1.f;
      pt[wave][r + 8 * hf][lr]      = (_Float16)(e0 * g);
      pt[wave][r + 8 * hf][16 + lr] = (_Float16)(e1 * g);
    }

    // acc += P(16x32) @ V_tile(32x128): 1 K-chunk x 8 d-subtiles.
    v16h pa = ldA(&pt[wave][lr][0], hf);
#pragma unroll
    for (int j = 0; j < 8; ++j) {
      v16h vb = ldB(&vtT[16 * j + lr][0], hf);
      acc[j] = WMMA_F16X32(pa, vb, acc[j]);
    }
  }

  // Normalize, write attention output [B,T,C] (c = h*128 + d), f32.
#pragma unroll
  for (int r = 0; r < 8; ++r) {
    const float inv = 1.f / lsum[r];
    const size_t base = ((size_t)(b * Tc + t0 + r + 8 * hf)) * Cc + h * Dh;
#pragma unroll
    for (int j = 0; j < 8; ++j) attn[base + 16 * j + lr] = acc[j][r] * inv;
  }
}

// ---------------------------------------------------------------------------
// Host-side orchestration.
// ---------------------------------------------------------------------------
extern "C" void kernel_launch(void* const* d_in, const int* in_sizes, int n_in,
                              void* d_out, int out_size, void* d_ws,
                              size_t ws_size, hipStream_t stream) {
  (void)in_sizes; (void)n_in; (void)out_size; (void)ws_size;

  const float* x    = (const float*)d_in[0];
  const float* fm   = (const float*)d_in[1];
  const float* rm   = (const float*)d_in[2];
  const float* ctrl = (const float*)d_in[3];
  const float* Wq   = (const float*)d_in[4];
  const float* Wk   = (const float*)d_in[5];
  const float* Wv   = (const float*)d_in[6];
  const float* Wo   = (const float*)d_in[7];
  const float* Wc   = (const float*)d_in[8];
  const float* Wg   = (const float*)d_in[9];
  const float* bg   = (const float*)d_in[10];
  float* out = (float*)d_out;

  char* w = (char*)d_ws;
  float*    qb    = (float*)w;     w += (size_t)Bc * Tc * Cc * 4;  // [B*T,C] f32
  _Float16* kb    = (_Float16*)w;  w += (size_t)Bc * Sc * Cc * 2;  // [B*S,C] f16
  _Float16* vb    = (_Float16*)w;  w += (size_t)Bc * Sc * Cc * 2;  // [B*S,C] f16
  float*    attnb = (float*)w;     w += (size_t)Bc * Tc * Cc * 4;  // [B*T,C] f32
  float*    gateb = (float*)w;     w += (size_t)Bc * Tc * Hc * 4;  // [B*T,H]
  float*    cw    = (float*)w;                                     // [C]

  // 1) ctrl @ Wc^T bias
  ctrl_bias_kernel<<<3, 256, 0, stream>>>(ctrl, Wc, cw);
  // 2) q = x @ Wq^T + cw   (f32 out)
  gemm_xwT_64x64<<<dim3(Bc * Tc / 64, Cc / 64), 128, 0, stream>>>(
      x, x, x, Tc, 0, 0, Wq, cw, qb, 0, Cc, Cc);
  // 3) k,v = [x|fm|rm] @ Wk^T / Wv^T   (f16 out for the attention core)
  gemm_xwT_64x64<<<dim3(Bc * Sc / 64, Cc / 64), 128, 0, stream>>>(
      x, fm, rm, Tc, Mc, Rc, Wk, nullptr, kb, 1, Cc, Cc);
  gemm_xwT_64x64<<<dim3(Bc * Sc / 64, Cc / 64), 128, 0, stream>>>(
      x, fm, rm, Tc, Mc, Rc, Wv, nullptr, vb, 1, Cc, Cc);
  // 4) per-head memory gate
  gate_kernel<<<Bc * Tc, 32, 0, stream>>>(qb, Wg, bg, gateb);
  // 5) gated flash attention
  flash_cma<<<dim3(Tc / 64, Hc, Bc), 128, 0, stream>>>(qb, kb, vb, gateb,
                                                       attnb);
  // 6) final projection (f32 out)
  gemm_xwT_64x64<<<dim3(Bc * Tc / 64, Cc / 64), 128, 0, stream>>>(
      attnb, attnb, attnb, Tc, 0, 0, Wo, nullptr, out, 0, Cc, Cc);
}
